// RelHetGraph_11974368821770
// MI455X (gfx1250) — matrix operand order repaired and checked by reading, not compiled, not run on hardware
//
#include <hip/hip_runtime.h>
#include <hip/hip_bf16.h>

// ---------------------------------------------------------------------------
// Types for WMMA (CDNA5 / gfx1250, wave32)
// ---------------------------------------------------------------------------
typedef __attribute__((ext_vector_type(16))) __bf16          v16bf;
typedef __attribute__((ext_vector_type(8)))  float           v8f;
typedef __attribute__((ext_vector_type(8)))  unsigned short  v8u16;

union FragU { v8u16 u[2]; v16bf b; };

static __device__ __forceinline__ unsigned short f32_to_bf16_rne(float f) {
    unsigned u = __float_as_uint(f);
    u += 0x7FFFu + ((u >> 16) & 1u);
    return (unsigned short)(u >> 16);
}

// pack 8 floats -> v8u16 of bf16
static __device__ __forceinline__ v8u16 pack8_bf16(float4 a, float4 b) {
    v8u16 r;
    r[0] = f32_to_bf16_rne(a.x); r[1] = f32_to_bf16_rne(a.y);
    r[2] = f32_to_bf16_rne(a.z); r[3] = f32_to_bf16_rne(a.w);
    r[4] = f32_to_bf16_rne(b.x); r[5] = f32_to_bf16_rne(b.y);
    r[6] = f32_to_bf16_rne(b.z); r[7] = f32_to_bf16_rne(b.w);
    return r;
}

// ---------------------------------------------------------------------------
// Problem constants (match reference)
// ---------------------------------------------------------------------------
#define N_SENT 20000
#define N_WORD 50000
#define NEDGE  500000
#define DIM    768
#define HID    512   // conv1: 8 heads x 64
#define OUTC   256   // conv2: 1 head  x 256

// ---------------------------------------------------------------------------
// GEMM:  C[M,N] = act(A[M,K](f32) @ W[K,N](as bf16 WT[N,K]) + bias)
// Block = 256 threads = 8 waves. Tile 128x128, K-step 32.
// Each wave computes a 16x128 band -> 8 x v_wmma_f32_16x16x32_bf16 per K-step
// (1 A fragment reused across 8 B fragments; B fragments double-buffered so
//  ds_load of fragment nt+1 issues before the WMMA of fragment nt).
// ---------------------------------------------------------------------------
__global__ __launch_bounds__(256) void k_gemm_bf16(
    const float* __restrict__ A, const unsigned short* __restrict__ BT,
    const float* __restrict__ bias, float* __restrict__ C,
    int M, int N, int K, int relu)
{
    __shared__ __align__(16) unsigned short As[128][40]; // [m][k], padded
    __shared__ __align__(16) unsigned short Bs[128][40]; // [n][k], padded

    const int tid  = threadIdx.x;
    const int lane = tid & 31;
    const int wave = tid >> 5;
    const int h    = lane >> 4;   // K-half selector per ISA 16-bit layout
    const int lm   = lane & 15;   // M (A frag) or N (B frag) index

    const int m0 = blockIdx.x * 128;
    const int n0 = blockIdx.y * 128;

    // staging coordinates (fixed per thread): 2 threads per row, 16-col chunks
    const int ar  = tid >> 1;            // A row 0..127
    const int ac0 = (tid & 1) * 16;      // A col chunk 0 or 16
    const int agr = m0 + ar;
    const int bn  = tid >> 1;            // B row (n) 0..127
    const int bc0 = (tid & 1) * 16;      // B col (k) chunk 0 or 16

    const float*          arow = A  + (size_t)(agr < M ? agr : 0) * K;
    const unsigned short* brow = BT + (size_t)(n0 + bn) * K;

    v8f acc[8] = {};

    for (int k0 = 0; k0 < K; k0 += 32) {
        // ---- stage A (convert f32 -> bf16), fully vectorized ----
        v8u16 alo = {}, ahi = {};
        if (agr < M) {
            const float4* ap = (const float4*)(arow + k0 + ac0);
            alo = pack8_bf16(ap[0], ap[1]);
            ahi = pack8_bf16(ap[2], ap[3]);
        }
        *(v8u16*)&As[ar][ac0]     = alo;
        *(v8u16*)&As[ar][ac0 + 8] = ahi;

        // ---- stage B (already bf16, transposed in global) ----
        *(v8u16*)&Bs[bn][bc0]     = *(const v8u16*)(brow + k0 + bc0);
        *(v8u16*)&Bs[bn][bc0 + 8] = *(const v8u16*)(brow + k0 + bc0 + 8);

        // prefetch next A K-slab
        if (k0 + 32 < K) __builtin_prefetch(arow + k0 + 32 + ac0, 0, 0);

        __syncthreads();

        // A fragment: row (wave*16+lm), K = {h*16+0..15}
        const int mr = wave * 16 + lm;
        FragU a;
        a.u[0] = *(const v8u16*)&As[mr][h * 16];
        a.u[1] = *(const v8u16*)&As[mr][h * 16 + 8];

        // B fragments double-buffered: issue loads for nt+1 before WMMA nt
        FragU bf[2];
        bf[0].u[0] = *(const v8u16*)&Bs[lm][h * 16];
        bf[0].u[1] = *(const v8u16*)&Bs[lm][h * 16 + 8];

#pragma unroll
        for (int nt = 0; nt < 8; ++nt) {
            const int cur = nt & 1, nxt = cur ^ 1;
            if (nt < 7) {
                const int nn = (nt + 1) * 16 + lm;
                bf[nxt].u[0] = *(const v8u16*)&Bs[nn][h * 16];
                bf[nxt].u[1] = *(const v8u16*)&Bs[nn][h * 16 + 8];
            }
            acc[nt] = __builtin_amdgcn_wmma_f32_16x16x32_bf16(
                false, a.b, false, bf[cur].b, (short)0, acc[nt], false, false);
        }
        __syncthreads();
    }

    // Epilogue: C/D layout -> lane[4] picks M+8, vgpr v = M row offset
#pragma unroll
    for (int nt = 0; nt < 8; ++nt) {
#pragma unroll
        for (int v = 0; v < 8; ++v) {
            int row = m0 + wave * 16 + h * 8 + v;
            int col = n0 + nt * 16 + lm;
            if (row < M) {
                float val = acc[nt][v];
                if (bias) val += bias[col];
                if (relu && val < 0.0f) val = 0.0f;
                C[(size_t)row * N + col] = val;
            }
        }
    }
}

// ---------------------------------------------------------------------------
// Elementwise helpers
// ---------------------------------------------------------------------------
// weights: f32 [K,N] -> bf16 transposed [N,K]
__global__ void k_w_to_bf16_t(const float* __restrict__ s,
                              unsigned short* __restrict__ d, int K, int N) {
    size_t i = (size_t)blockIdx.x * blockDim.x + threadIdx.x;
    if (i >= (size_t)K * N) return;
    int k = (int)(i / N), n = (int)(i % N);
    d[(size_t)n * K + k] = f32_to_bf16_rne(s[i]);
}

__global__ void k_fill_u32(unsigned* __restrict__ p, unsigned v, size_t n) {
    size_t i = (size_t)blockIdx.x * blockDim.x + threadIdx.x;
    if (i < n) p[i] = v;
}

// out[n, :] = b1 + b2   (sum of the two relations' biases feeding this dst type)
__global__ void k_bias_init(float* __restrict__ out, const float* __restrict__ b1,
                            const float* __restrict__ b2, size_t Nn, int Ct) {
    size_t i = (size_t)blockIdx.x * blockDim.x + threadIdx.x;
    if (i < Nn * (size_t)Ct) {
        int c = (int)(i % Ct);
        out[i] = b1[c] + b2[c];
    }
}

// al[n,h] = sum_c x[n, h*C + c] * att[h*C + c]
__global__ void k_att_logits(const float* __restrict__ x, const float* __restrict__ att,
                             float* __restrict__ al, int Nn, int H, int C) {
    size_t i = (size_t)blockIdx.x * blockDim.x + threadIdx.x;
    if (i >= (size_t)Nn * H) return;
    int n = (int)(i / H), h = (int)(i % H);
    const float4* xr = (const float4*)(x   + (size_t)n * H * C + (size_t)h * C);
    const float4* ar = (const float4*)(att + (size_t)h * C);
    float s = 0.0f;
    for (int c = 0; c < C / 4; ++c) {
        float4 xv = xr[c], av = ar[c];
        s += xv.x * av.x + xv.y * av.y + xv.z * av.z + xv.w * av.w;
    }
    al[i] = s;
}

// ---------------------------------------------------------------------------
// Edge passes.  Order-preserving float<->uint encoding for atomic max.
// ---------------------------------------------------------------------------
static __device__ __forceinline__ unsigned enc_f32(float f) {
    unsigned u = __float_as_uint(f);
    return u ^ ((u & 0x80000000u) ? 0xFFFFFFFFu : 0x80000000u);
}
static __device__ __forceinline__ float dec_f32(unsigned u) {
    return __uint_as_float(u ^ ((u & 0x80000000u) ? 0x80000000u : 0xFFFFFFFFu));
}
static __device__ __forceinline__ float lrelu02(float v) {
    return v < 0.0f ? 0.2f * v : v;
}

__global__ void k_edge_max(const int* __restrict__ src, const int* __restrict__ dst,
                           const float* __restrict__ als, const float* __restrict__ ald,
                           unsigned* __restrict__ mu, int E, int H) {
    size_t i = (size_t)blockIdx.x * blockDim.x + threadIdx.x;
    if (i >= (size_t)E * H) return;
    int e = (int)(i / H), h = (int)(i % H);
    int s = src[e], d = dst[e];
    float v = lrelu02(als[(size_t)s * H + h] + ald[(size_t)d * H + h]);
    atomicMax(&mu[(size_t)d * H + h], enc_f32(v));
}

__global__ void k_edge_exp(const int* __restrict__ src, const int* __restrict__ dst,
                           const float* __restrict__ als, const float* __restrict__ ald,
                           const unsigned* __restrict__ mu, float* __restrict__ ex,
                           float* __restrict__ den, int E, int H) {
    size_t i = (size_t)blockIdx.x * blockDim.x + threadIdx.x;
    if (i >= (size_t)E * H) return;
    int e = (int)(i / H), h = (int)(i % H);
    int s = src[e], d = dst[e];
    float v = lrelu02(als[(size_t)s * H + h] + ald[(size_t)d * H + h]);
    float m = dec_f32(mu[(size_t)d * H + h]);
    float ev = __expf(v - m);
    ex[i] = ev;
    atomicAdd(&den[(size_t)d * H + h], ev);
}

// one block per edge: out[dst, c] += (ex/den) * xs[src, c]
__global__ __launch_bounds__(256) void k_edge_scatter(
    const int* __restrict__ src, const int* __restrict__ dst,
    const float* __restrict__ ex, const float* __restrict__ den,
    const float* __restrict__ xs, float* __restrict__ out,
    int E, int H, int C) {
    int e = blockIdx.x;
    if (e >= E) return;
    int s = src[e], d = dst[e];
    int Ct = H * C;
    const float* xrow = xs  + (size_t)s * Ct;
    float*       orow = out + (size_t)d * Ct;
    for (int c = threadIdx.x; c < Ct; c += blockDim.x) {
        int hh = c / C;
        float alpha = ex[(size_t)e * H + hh] / den[(size_t)d * H + hh];
        atomicAdd(&orow[c], alpha * xrow[c]);
    }
}

// ---------------------------------------------------------------------------
// Host orchestration
// ---------------------------------------------------------------------------
extern "C" void kernel_launch(void* const* d_in, const int* in_sizes, int n_in,
                              void* d_out, int out_size, void* d_ws, size_t ws_size,
                              hipStream_t stream) {
    // ---- inputs (setup_inputs dict order, nested structures flattened) ----
    const float* sent   = (const float*)d_in[0];
    const float* word   = (const float*)d_in[1];
    const int *es[4], *ed[4];                 // sim, pa, in, has
    for (int i = 0; i < 4; ++i) { es[i] = (const int*)d_in[2 + 2*i]; ed[i] = (const int*)d_in[3 + 2*i]; }
    const float* W_sent = (const float*)d_in[10];
    const float* b_sent = (const float*)d_in[11];
    const float* W_word = (const float*)d_in[12];
    const float* b_word = (const float*)d_in[13];
    const float *c1W[4], *c1as[4], *c1ad[4], *c1b[4];
    for (int i = 0; i < 4; ++i) {
        c1W[i]  = (const float*)d_in[14 + 4*i];
        c1as[i] = (const float*)d_in[15 + 4*i];
        c1ad[i] = (const float*)d_in[16 + 4*i];
        c1b[i]  = (const float*)d_in[17 + 4*i];
    }
    // conv2: sim @30..33, pa @34..37, in @38..41, has @42..45 (only sim & in used)
    const float* c2W_sim  = (const float*)d_in[30];
    const float* c2as_sim = (const float*)d_in[31];
    const float* c2ad_sim = (const float*)d_in[32];
    const float* c2b_sim  = (const float*)d_in[33];
    const float* c2W_in   = (const float*)d_in[38];
    const float* c2as_in  = (const float*)d_in[39];
    const float* c2ad_in  = (const float*)d_in[40];
    const float* c2b_in   = (const float*)d_in[41];

    // ---- workspace carve ----
    char*  base = (char*)d_ws;
    size_t off  = 0;
    auto carve = [&](size_t bytes) -> void* {
        void* p = base + off;
        off = (off + bytes + 255) & ~(size_t)255;
        return p;
    };
    unsigned short* wS   = (unsigned short*)carve((size_t)DIM * DIM * 2);
    unsigned short* wW   = (unsigned short*)carve((size_t)DIM * DIM * 2);
    unsigned short* w1bf[4];
    for (int i = 0; i < 4; ++i) w1bf[i] = (unsigned short*)carve((size_t)DIM * HID * 2);
    unsigned short* w2sim = (unsigned short*)carve((size_t)HID * OUTC * 2);
    unsigned short* w2in  = (unsigned short*)carve((size_t)HID * OUTC * 2);
    float* hs    = (float*)carve((size_t)N_SENT * DIM * 4);
    float* hw    = (float*)carve((size_t)N_WORD * DIM * 4);
    float* projA = (float*)carve((size_t)N_WORD * HID * 4);  // big (src/dst up to 50000)
    float* projB = (float*)carve((size_t)N_SENT * HID * 4);  // small (<=20000 rows)
    float* al_s  = (float*)carve((size_t)N_WORD * 8 * 4);
    float* al_d  = (float*)carve((size_t)N_WORD * 8 * 4);
    unsigned* mu = (unsigned*)carve((size_t)N_WORD * 8 * 4);
    float* den   = (float*)carve((size_t)N_WORD * 8 * 4);
    float* ex    = (float*)carve((size_t)NEDGE * 8 * 4);
    float* s1    = (float*)carve((size_t)N_SENT * HID * 4);
    float* w1    = (float*)carve((size_t)N_WORD * HID * 4);
    (void)ws_size; (void)n_in; (void)in_sizes; (void)out_size;

    // ---- launch helpers ----
    auto cvtT = [&](const float* s, unsigned short* d, int K, int N) {
        size_t n = (size_t)K * N;
        k_w_to_bf16_t<<<(unsigned)((n + 255) / 256), 256, 0, stream>>>(s, d, K, N);
    };
    auto gemm = [&](const float* A, const unsigned short* BT, const float* bias,
                    float* C, int M, int N, int K, int relu) {
        dim3 g((M + 127) / 128, N / 128);
        k_gemm_bf16<<<g, 256, 0, stream>>>(A, BT, bias, C, M, N, K, relu);
    };
    auto fill = [&](void* p, unsigned v, size_t n) {
        k_fill_u32<<<(unsigned)((n + 255) / 256), 256, 0, stream>>>((unsigned*)p, v, n);
    };
    auto logits = [&](const float* x, const float* att, float* al, int Nn, int H, int C) {
        size_t t = (size_t)Nn * H;
        k_att_logits<<<(unsigned)((t + 255) / 256), 256, 0, stream>>>(x, att, al, Nn, H, C);
    };
    auto relation = [&](const float* xsrc, int Ns, const float* xdst, int Nd,
                        const unsigned short* WbfT, const float* atts, const float* attd,
                        const int* srcI, const int* dstI,
                        float* projS, float* projD, float* outAcc,
                        int H, int C, int Kin) {
        gemm(xsrc, WbfT, nullptr, projS, Ns, H * C, Kin, 0);
        if (projD != projS) gemm(xdst, WbfT, nullptr, projD, Nd, H * C, Kin, 0);
        logits(projS, atts, al_s, Ns, H, C);
        logits(projD, attd, al_d, Nd, H, C);
        fill(mu, 0u, (size_t)Nd * H);
        fill(den, 0u, (size_t)Nd * H);
        size_t teh = (size_t)NEDGE * H;
        unsigned gb = (unsigned)((teh + 255) / 256);
        k_edge_max<<<gb, 256, 0, stream>>>(srcI, dstI, al_s, al_d, mu, NEDGE, H);
        k_edge_exp<<<gb, 256, 0, stream>>>(srcI, dstI, al_s, al_d, mu, ex, den, NEDGE, H);
        k_edge_scatter<<<NEDGE, 256, 0, stream>>>(srcI, dstI, ex, den, projS, outAcc, NEDGE, H, C);
    };

    // ---- weight conversion (f32 [K,N] -> bf16 transposed [N,K]) ----
    cvtT(W_sent, wS, DIM, DIM);
    cvtT(W_word, wW, DIM, DIM);
    for (int i = 0; i < 4; ++i) cvtT(c1W[i], w1bf[i], DIM, HID);
    cvtT(c2W_sim, w2sim, HID, OUTC);
    cvtT(c2W_in,  w2in,  HID, OUTC);

    // ---- dense feature transforms: relu(x @ W + b) ----
    gemm(sent, wS, b_sent, hs, N_SENT, DIM, DIM, 1);
    gemm(word, wW, b_word, hw, N_WORD, DIM, DIM, 1);

    // ---- conv1 (HeteroConv, sum over relations per dst type) ----
    {   // dst accumulators start at the sum of the contributing biases
        size_t t1 = (size_t)N_SENT * HID;
        k_bias_init<<<(unsigned)((t1 + 255) / 256), 256, 0, stream>>>(s1, c1b[0], c1b[2], N_SENT, HID);
        size_t t2 = (size_t)N_WORD * HID;
        k_bias_init<<<(unsigned)((t2 + 255) / 256), 256, 0, stream>>>(w1, c1b[1], c1b[3], N_WORD, HID);
    }
    // sim: sent->sent (shared projection)
    relation(hs, N_SENT, hs, N_SENT, w1bf[0], c1as[0], c1ad[0], es[0], ed[0],
             projA, projA, s1, 8, 64, DIM);
    // in: word->sent
    relation(hw, N_WORD, hs, N_SENT, w1bf[2], c1as[2], c1ad[2], es[2], ed[2],
             projA, projB, s1, 8, 64, DIM);
    // pa: sent->word
    relation(hs, N_SENT, hw, N_WORD, w1bf[1], c1as[1], c1ad[1], es[1], ed[1],
             projB, projA, w1, 8, 64, DIM);
    // has: sent->word
    relation(hs, N_SENT, hw, N_WORD, w1bf[3], c1as[3], c1ad[3], es[3], ed[3],
             projB, projA, w1, 8, 64, DIM);

    // ---- conv2 (only sentence-dst relations matter) ----
    float* outf = (float*)d_out;
    {
        size_t t = (size_t)N_SENT * OUTC;
        k_bias_init<<<(unsigned)((t + 255) / 256), 256, 0, stream>>>(outf, c2b_sim, c2b_in, N_SENT, OUTC);
    }
    // sim: s1 -> s2 (shared projection)
    relation(s1, N_SENT, s1, N_SENT, w2sim, c2as_sim, c2ad_sim, es[0], ed[0],
             projA, projA, outf, 1, OUTC, HID);
    // in: w1 -> s2
    relation(w1, N_WORD, s1, N_SENT, w2in, c2as_in, c2ad_in, es[2], ed[2],
             projA, projB, outf, 1, OUTC, HID);
}
